// HybridQLSTM_65481071409418
// MI455X (gfx1250) — matrix operand order repaired and checked by reading, not compiled, not run on hardware
//
#include <hip/hip_runtime.h>
#include <hip/hip_bf16.h>
#include <math.h>

#define S_LEN 4096
#define E_DIM 1024
#define H_DIM 1024
#define G_DIM 4096   // 4H
#define T_DIM 1024

typedef __bf16 bf16_t;
typedef bf16_t v16bf __attribute__((ext_vector_type(16)));
typedef float  v8f   __attribute__((ext_vector_type(8)));

// ---- async global->LDS helpers (CDNA5): 16B per lane, tracked by ASYNCcnt ----
__device__ __forceinline__ void async_g2l_b128(const void* gsrc, const void* ldst) {
  unsigned loff = (unsigned)(uintptr_t)ldst;   // LDS aperture: addr[31:0] == LDS byte offset
  asm volatile("global_load_async_to_lds_b128 %0, %1, off"
               :: "v"(loff), "v"(gsrc) : "memory");
}
__device__ __forceinline__ void wait_async0() {
#if defined(__has_builtin) && __has_builtin(__builtin_amdgcn_s_wait_asynccnt)
  __builtin_amdgcn_s_wait_asynccnt(0);
#else
  asm volatile("s_wait_asynccnt 0" ::: "memory");
#endif
}

// ---------------- init: zero barrier counter + h double buffers ----------------
__global__ void k_init(unsigned* ctr, float* hb0, float* hb1) {
  if (blockIdx.x == 0 && threadIdx.x == 0) *ctr = 0u;
  for (int i = blockIdx.x * blockDim.x + threadIdx.x; i < H_DIM; i += blockDim.x * gridDim.x) {
    hb0[i] = 0.f; hb1[i] = 0.f;
  }
}

// ---------------- convert W_ih to bf16 ----------------
__global__ void k_cvt(const float* __restrict__ w, bf16_t* __restrict__ wb, int n) {
  int idx = blockIdx.x * blockDim.x + threadIdx.x;
  if (idx < n) wb[idx] = (bf16_t)w[idx];
}

// ---------------- embedding gather + bf16 convert ----------------
__global__ void k_gather(const int* __restrict__ sent, const float* __restrict__ emb,
                         bf16_t* __restrict__ xb) {
  size_t idx = (size_t)blockIdx.x * blockDim.x + threadIdx.x;  // covers S*E
  int s = (int)(idx >> 10);
  int k = (int)(idx & 1023);
  xb[idx] = (bf16_t)emb[(size_t)sent[s] * E_DIM + k];
}

// ---------------- x_gates = x @ W_ih^T + (b_ih + b_hh)  (WMMA bf16) ----------------
// block: 256 threads = 8 waves; block tile 32(M) x 256(N); wave tile 16(M) x 64(N)
__global__ __launch_bounds__(256)
void k_gemm_xgates(const bf16_t* __restrict__ A,   // [S, E] bf16 (gathered embeddings)
                   const bf16_t* __restrict__ W,   // [4H, E] bf16
                   const float* __restrict__ b_ih,
                   const float* __restrict__ b_hh,
                   float* __restrict__ C)          // [S, 4H] f32
{
  const int lane = threadIdx.x & 31;
  const int wave = threadIdx.x >> 5;   // 0..7
  const int wm   = wave >> 2;          // 0..1
  const int wn   = wave & 3;           // 0..3
  const int m0   = blockIdx.y * 32 + wm * 16;
  const int n0   = blockIdx.x * 256 + wn * 64;
  const int half = lane >> 4;
  const int l15  = lane & 15;

  v8f acc[4] = {};

  // A-frag per-lane base: row (m0+l15), K offset half*8 (elems 0..7) and +16 (elems 8..15)
  const bf16_t* arow = A + (size_t)(m0 + l15) * E_DIM + half * 8;

  for (int k0 = 0; k0 < E_DIM; k0 += 32) {
    union { v16bf v; float4 f[2]; } ua;
    ua.f[0] = *(const float4*)(arow + k0);
    ua.f[1] = *(const float4*)(arow + k0 + 16);
#pragma unroll
    for (int j = 0; j < 4; ++j) {
      // B-frag: lane = column n, K offset half*8 (elems 0..7) and +16 (elems 8..15)
      const bf16_t* brow = W + (size_t)(n0 + j * 16 + l15) * E_DIM + k0 + half * 8;
      union { v16bf v; float4 f[2]; } ub;
      ub.f[0] = *(const float4*)brow;
      ub.f[1] = *(const float4*)(brow + 16);
      acc[j] = __builtin_amdgcn_wmma_f32_16x16x32_bf16(
          false, ua.v, false, ub.v, (short)0, acc[j], false, false);
    }
  }

  // C/D layout: VGPR v -> M = m0 + half*8 + v, N = n0 + l15
  const int mb = m0 + half * 8;
#pragma unroll
  for (int j = 0; j < 4; ++j) {
    int n = n0 + j * 16 + l15;
    float bias = b_ih[n] + b_hh[n];
#pragma unroll
    for (int v = 0; v < 8; ++v)
      C[(size_t)(mb + v) * G_DIM + n] = acc[j][v] + bias;
  }
}

// ---------------- persistent LSTM recurrence ----------------
// 64 WGs x 256 threads. WG wg owns hidden units [wg*16, wg*16+16) -> 64 W_hh rows
// resident in LDS. Row layout is quarter-blocked+padded so the dot loop runs on
// conflict-free ds_load_b128: dword offset = r*RSTRIDE + q*QSTR + k, bank(16B chunk)
// = 16r + 4q, distinct for all 32 lanes of a wave.
#define RWGS   64
#define JPW    16            // hidden units per WG
#define NROWS  64            // 4 gates * JPW
#define QSTR   260           // dwords per quarter (256 + 4 pad)
#define RSTRIDE (4 * QSTR)   // 1040 dwords per row

__global__ __launch_bounds__(256)
void k_lstm(const float* __restrict__ xg,    // [S, 4H]
            const float* __restrict__ Whh,   // [4H, H]
            float* __restrict__ hs,          // [S, H]
            float* __restrict__ hb0, float* __restrict__ hb1,
            unsigned* __restrict__ ctr)
{
  extern __shared__ float rows[];            // NROWS * RSTRIDE dwords
  __shared__ float hsh[4 * QSTR];            // h staged, quarter-padded
  __shared__ float gred[NROWS];
  __shared__ float cst[JPW];

  const int wg = blockIdx.x;
  const int t  = threadIdx.x;

  // stage this WG's 64 W_hh rows into LDS via async tensor path (no VGPR round trip)
  {
    const int k = 4 * t;                     // each thread: one 16B chunk per row
    const int ldd = (k >> 8) * QSTR + (k & 255);
    for (int r = 0; r < NROWS; ++r) {
      int gate = r >> 4, jj = r & 15;
      const float* src = Whh + (size_t)(gate * H_DIM + wg * JPW + jj) * H_DIM + k;
      async_g2l_b128(src, &rows[r * RSTRIDE + ldd]);
    }
  }
  if (t < JPW) cst[t] = 0.f;
  wait_async0();
  __syncthreads();

  const int r    = t >> 2;          // 0..63: row handled by this thread
  const int q    = t & 3;           // quarter of K handled by this thread
  const int gate = r >> 4;
  const int jj   = r & 15;
  const int gidx = gate * H_DIM + wg * JPW + jj;
  const float* wq = rows + r * RSTRIDE + q * QSTR;
  const float* hq = hsh + q * QSTR;

  float* hb[2] = { hb0, hb1 };
  const int hk  = 4 * t;                               // this thread's h-chunk
  const int hkd = (hk >> 8) * QSTR + (hk & 255);       // padded LDS dword offset

  for (int step = 0; step < S_LEN; ++step) {
    const float* hread = hb[step & 1];
    // stage h into LDS directly from L2 via async load (also avoids stale-L0 reads)
    async_g2l_b128(hread + hk, &hsh[hkd]);
    wait_async0();
    __syncthreads();

    // dot(row_quarter, h_quarter): b128 LDS reads, weights bank-disjoint, h broadcast
    float ax = 0.f, ay = 0.f, az = 0.f, aw = 0.f;
#pragma unroll 8
    for (int kk = 0; kk < 64; ++kk) {
      float4 w4 = *(const float4*)(wq + 4 * kk);
      float4 h4 = *(const float4*)(hq + 4 * kk);
      ax = fmaf(w4.x, h4.x, ax);
      ay = fmaf(w4.y, h4.y, ay);
      az = fmaf(w4.z, h4.z, az);
      aw = fmaf(w4.w, h4.w, aw);
    }
    float acc = (ax + ay) + (az + aw);
    acc += __shfl_xor(acc, 1);
    acc += __shfl_xor(acc, 2);
    if (q == 0) gred[r] = acc + xg[(size_t)step * G_DIM + gidx];
    __syncthreads();

    if (t < JPW) {
      float gi = gred[t], gf = gred[16 + t], gg = gred[32 + t], go = gred[48 + t];
      float i = 1.f / (1.f + __expf(-gi));
      float f = 1.f / (1.f + __expf(-gf));
      float g = tanhf(gg);
      float o = 1.f / (1.f + __expf(-go));
      float c = f * cst[t] + i * g;
      cst[t] = c;
      float hn = o * tanhf(c);
      int hidx = wg * JPW + t;
      hs[(size_t)step * H_DIM + hidx] = hn;
      hb[(step + 1) & 1][hidx] = hn;   // write into the *other* buffer (no WAR race)
    }

    // device-visible release of h writes, then grid barrier (monotonic counter)
    __threadfence();
    __syncthreads();
    if (t == 0) {
      __hip_atomic_fetch_add(ctr, 1u, __ATOMIC_ACQ_REL, __HIP_MEMORY_SCOPE_AGENT);
      unsigned target = (unsigned)(step + 1) * RWGS;
      while (__hip_atomic_load(ctr, __ATOMIC_ACQUIRE, __HIP_MEMORY_SCOPE_AGENT) < target)
        __builtin_amdgcn_s_sleep(1);
    }
    __syncthreads();
  }
}

// ---------------- collapsed head: v = W_hyb @ W_tag, c0 = W_hyb·b_tag + b_hyb ----------------
__global__ void k_head_v(const float* __restrict__ Wtag, const float* __restrict__ btag,
                         const float* __restrict__ Whyb, const float* __restrict__ bhyb,
                         float* __restrict__ v, float* __restrict__ c0)
{
  int h = blockIdx.x * blockDim.x + threadIdx.x;
  if (h < H_DIM) {
    float a = 0.f;
    for (int k = 0; k < T_DIM; ++k) a = fmaf(Whyb[k], Wtag[(size_t)k * H_DIM + h], a);
    v[h] = a;
  }
  if (h == 0) {
    float a = 0.f;
    for (int k = 0; k < T_DIM; ++k) a = fmaf(Whyb[k], btag[k], a);
    *c0 = a + bhyb[0];
  }
}

// out[s] = sigmoid(hs[s,:]·v + c0); one wave per row
__global__ __launch_bounds__(256)
void k_head_out(const float* __restrict__ hs, const float* __restrict__ v,
                const float* __restrict__ c0, float* __restrict__ out)
{
  int gtid = blockIdx.x * blockDim.x + threadIdx.x;
  int row  = gtid >> 5;
  int lane = gtid & 31;
  if (row >= S_LEN) return;
  const float* hr = hs + (size_t)row * H_DIM;
  float a = 0.f;
  for (int k = lane; k < H_DIM; k += 32) a = fmaf(hr[k], v[k], a);
  a += __shfl_xor(a, 16); a += __shfl_xor(a, 8); a += __shfl_xor(a, 4);
  a += __shfl_xor(a, 2);  a += __shfl_xor(a, 1);
  if (lane == 0) out[row] = 1.f / (1.f + __expf(-(a + *c0)));
}

// ---------------- host launcher ----------------
extern "C" void kernel_launch(void* const* d_in, const int* in_sizes, int n_in,
                              void* d_out, int out_size, void* d_ws, size_t ws_size,
                              hipStream_t stream)
{
  const int*   sent = (const int*)  d_in[0];
  const float* emb  = (const float*)d_in[1];
  const float* Wih  = (const float*)d_in[2];
  const float* Whh  = (const float*)d_in[3];
  const float* bih  = (const float*)d_in[4];
  const float* bhh  = (const float*)d_in[5];
  const float* Wtag = (const float*)d_in[6];
  const float* btag = (const float*)d_in[7];
  const float* Whyb = (const float*)d_in[8];
  const float* bhyb = (const float*)d_in[9];
  float* out = (float*)d_out;

  // workspace carve-out (256B aligned)
  char* ws = (char*)d_ws;
  size_t off = 0;
  auto take = [&](size_t bytes) -> char* {
    char* p = ws + off;
    off = (off + bytes + 255) & ~(size_t)255;
    return p;
  };
  unsigned* ctr = (unsigned*)take(sizeof(unsigned));
  float*  hb0 = (float*) take((size_t)H_DIM * 4);
  float*  hb1 = (float*) take((size_t)H_DIM * 4);
  float*  v   = (float*) take((size_t)H_DIM * 4);
  float*  c0  = (float*) take(4);
  bf16_t* xb  = (bf16_t*)take((size_t)S_LEN * E_DIM * 2);
  bf16_t* wb  = (bf16_t*)take((size_t)G_DIM * E_DIM * 2);
  float*  xg  = (float*) take((size_t)S_LEN * G_DIM * 4);
  float*  hs  = (float*) take((size_t)S_LEN * H_DIM * 4);

  k_init<<<4, 256, 0, stream>>>(ctr, hb0, hb1);
  k_cvt<<<(G_DIM * E_DIM + 255) / 256, 256, 0, stream>>>(Wih, wb, G_DIM * E_DIM);
  k_gather<<<(S_LEN * E_DIM) / 256, 256, 0, stream>>>(sent, emb, xb);

  dim3 gg(G_DIM / 256, S_LEN / 32);
  k_gemm_xgates<<<gg, 256, 0, stream>>>(xb, wb, bih, bhh, xg);

  size_t lds = (size_t)NROWS * RSTRIDE * sizeof(float);  // 266240 B (<320KB WGP LDS)
  (void)hipFuncSetAttribute(reinterpret_cast<const void*>(k_lstm),
                            hipFuncAttributeMaxDynamicSharedMemorySize, (int)lds);
  k_lstm<<<RWGS, 256, lds, stream>>>(xg, Whh, hs, hb0, hb1, ctr);

  k_head_v<<<(H_DIM + 255) / 256, 256, 0, stream>>>(Wtag, btag, Whyb, bhyb, v, c0);
  k_head_out<<<(S_LEN * 32) / 256, 256, 0, stream>>>(hs, v, c0, out);
}
